// GAAE_mod3_66657892434574
// MI455X (gfx1250) — matrix operand order repaired and checked by reading, not compiled, not run on hardware
//
#include <hip/hip_runtime.h>
#include <hip/hip_bf16.h>

// ---------------------------------------------------------------------------
// GAAE / STAGATE forward on gfx1250 (MI455X).
// Dominant cost: 2 x 153.6 GFLOP GEMMs -> v_wmma_f32_16x16x32_bf16 (fp32 acc).
// Roofline: 600MB feature read / 600MB h4 write ~= 26us each @ 23.3TB/s; bf16
// WMMA puts GEMM compute at the same order (fp32 16x16x4 would be ~8x worse).
// This rev: guard-free interior-block bodies (block-uniform dispatch) and a
// split K loop (full tiles + one guarded epilogue), so the hot loop is purely
// b128 loads -> bf16 pack -> LDS -> 12x ds_load_b128 -> 8x WMMA per K-step.
// Decoder A tiles move via global_load_async_to_lds_b128 (ASYNCcnt).
// Edge softmax: logits = sigmoid(.) in (0,1) -> exp stable, segment-max pass
// dropped (weights mathematically identical; denom >= e^-1 >> 1e-16).
// ---------------------------------------------------------------------------

typedef __attribute__((ext_vector_type(16))) __bf16        v16bf;
typedef __attribute__((ext_vector_type(8)))  float         v8f;
typedef __attribute__((ext_vector_type(4)))  float         f32x4;
typedef __attribute__((ext_vector_type(2)))  unsigned int  u32x2;

static constexpr int Gd = 3000;  // genes (K of GEMM1, N of GEMM4)
static constexpr int Hd = 512;   // hidden
static constexpr int Zd = 30;    // latent
static constexpr int Pd = 20;    // pred classes

// float -> bf16 RNE (finite inputs)
static __device__ inline unsigned short f2bf(float f) {
  union { float f; unsigned int u; } x; x.f = f;
  unsigned int r = x.u + 0x7FFFu + ((x.u >> 16) & 1u);
  return (unsigned short)(r >> 16);
}
// pack bf16(a) into [15:0], bf16(b) into [31:16]
static __device__ inline unsigned int pack2bf(float a, float b) {
  union { float f; unsigned int u; } x, y; x.f = a; y.f = b;
  unsigned int ra = x.u + 0x7FFFu + ((x.u >> 16) & 1u);
  unsigned int rb = y.u + 0x7FFFu + ((y.u >> 16) & 1u);
  return (ra >> 16) | (rb & 0xFFFF0000u);
}

// Build a v16bf WMMA fragment from two 16-byte LDS chunks (2x ds_load_b128).
static __device__ inline v16bf ld_frag(const unsigned short* p0,
                                       const unsigned short* p1) {
  union { f32x4 q[2]; v16bf v; } u;
  u.q[0] = *reinterpret_cast<const f32x4*>(p0);
  u.q[1] = *reinterpret_cast<const f32x4*>(p1);
  return u.v;
}

#define WMMA_BF16(A_, B_, C_) \
  __builtin_amdgcn_wmma_f32_16x16x32_bf16(false, (A_), false, (B_), (short)0, (C_), false, false)

// LDS row stride (bf16 elems): 40 -> 80B rows, 16B-aligned chunks, spread banks.
static constexpr int LSTR = 40;

// Fragment loads + 8 WMMAs for one 128x128x32 tile step (per wave: 32x64).
static __device__ inline void wmma_tile_step(const unsigned short* sA,
                                             const unsigned short* sB,
                                             int wm, int wn, int r, int hh,
                                             v8f acc[2][4]) {
  v16bf a0 = ld_frag(&sA[(wm +      r) * LSTR + hh * 16], &sA[(wm +      r) * LSTR + hh * 16 + 8]);
  v16bf a1 = ld_frag(&sA[(wm + 16 + r) * LSTR + hh * 16], &sA[(wm + 16 + r) * LSTR + hh * 16 + 8]);
  v16bf b0 = ld_frag(&sB[(wn +      r) * LSTR + hh * 16], &sB[(wn +      r) * LSTR + hh * 16 + 8]);
  v16bf b1 = ld_frag(&sB[(wn + 16 + r) * LSTR + hh * 16], &sB[(wn + 16 + r) * LSTR + hh * 16 + 8]);
  v16bf b2 = ld_frag(&sB[(wn + 32 + r) * LSTR + hh * 16], &sB[(wn + 32 + r) * LSTR + hh * 16 + 8]);
  v16bf b3 = ld_frag(&sB[(wn + 48 + r) * LSTR + hh * 16], &sB[(wn + 48 + r) * LSTR + hh * 16 + 8]);
  acc[0][0] = WMMA_BF16(a0, b0, acc[0][0]);
  acc[0][1] = WMMA_BF16(a0, b1, acc[0][1]);
  acc[0][2] = WMMA_BF16(a0, b2, acc[0][2]);
  acc[0][3] = WMMA_BF16(a0, b3, acc[0][3]);
  acc[1][0] = WMMA_BF16(a1, b0, acc[1][0]);
  acc[1][1] = WMMA_BF16(a1, b1, acc[1][1]);
  acc[1][2] = WMMA_BF16(a1, b2, acc[1][2]);
  acc[1][3] = WMMA_BF16(a1, b3, acc[1][3]);
}

// D layout: VGPR v, lanes0-15 -> M=v, lanes16-31 -> M=v+8; N = lane&15.
template <bool GUARD>
static __device__ inline void store_tiles(float* __restrict__ C, int M, int NN,
                                          int m0, int n0, int wm, int wn,
                                          int r, int hh, v8f acc[2][4]) {
  for (int i = 0; i < 2; ++i)
    for (int j = 0; j < 4; ++j) {
      int col = n0 + wn + j * 16 + r;
      if (GUARD && col >= NN) continue;
      for (int v = 0; v < 8; ++v) {
        int row = m0 + wm + i * 16 + hh * 8 + v;
        if (!GUARD || row < M) C[(size_t)row * NN + col] = acc[i][j][v];
      }
    }
}

// ---------------------------------------------------------------------------
// GEMM1 body: C[M,NN] = (A[M,K]*gate[K]) @ B[K,NN]. 256 thr, 128x128, BK=32.
// 8 waves in 4(m) x 2(n); each wave owns 32x64 = 2x4 WMMA tiles.
// ---------------------------------------------------------------------------
template <bool GUARD>
static __device__ inline void gemm1_body(const float* __restrict__ A,
                                         const float* __restrict__ gate,
                                         const float* __restrict__ B,
                                         float* __restrict__ C,
                                         int M, int K, int NN, int m0, int n0) {
  __shared__ __align__(16) unsigned short sA[128 * LSTR];
  __shared__ __align__(16) unsigned short sB[128 * LSTR];  // B^T: [n][k]
  const int tid = threadIdx.x, lane = tid & 31, wave = tid >> 5;
  const int wm = (wave & 3) * 32, wn = (wave >> 2) * 64;
  const int r = lane & 15, hh = lane >> 4;
  v8f acc[2][4] = {};

  const int Kmain = K & ~31;
  for (int k0 = 0; k0 < Kmain; k0 += 32) {
    // ---- stage A: 128 rows x 32 k as 4-wide groups (no k guard in main loop)
    for (int i = 0; i < 4; ++i) {
      int idx = tid + i * 256;
      int rr = idx >> 3, g = idx & 7;
      int gm = m0 + rr, gk = k0 + g * 4;
      unsigned p0 = 0, p1 = 0;
      if (!GUARD || gm < M) {
        f32x4 va = *reinterpret_cast<const f32x4*>(A + (size_t)gm * K + gk);
        f32x4 vg = *reinterpret_cast<const f32x4*>(gate + gk);
        p0 = pack2bf(va[0] * vg[0], va[1] * vg[1]);
        p1 = pack2bf(va[2] * vg[2], va[3] * vg[3]);
      }
      u32x2 t; t[0] = p0; t[1] = p1;
      *reinterpret_cast<u32x2*>(&sA[rr * LSTR + g * 4]) = t;
    }
    // ---- stage B^T: 32k x 128n; each thread does 4k x 4n (one iter)
    {
      int n4 = tid & 31, kq = tid >> 5;
      int gn = n0 + n4 * 4, gk = k0 + kq * 4;
      f32x4 r0 = {}, r1 = {}, r2 = {}, r3 = {};
      if (!GUARD || gn + 3 < NN) {
        r0 = *reinterpret_cast<const f32x4*>(B + (size_t)(gk + 0) * NN + gn);
        r1 = *reinterpret_cast<const f32x4*>(B + (size_t)(gk + 1) * NN + gn);
        r2 = *reinterpret_cast<const f32x4*>(B + (size_t)(gk + 2) * NN + gn);
        r3 = *reinterpret_cast<const f32x4*>(B + (size_t)(gk + 3) * NN + gn);
      } else {
        for (int j = 0; j < 4; ++j) {
          int cn = gn + j;
          if (cn < NN) {
            r0[j] = B[(size_t)(gk + 0) * NN + cn];
            r1[j] = B[(size_t)(gk + 1) * NN + cn];
            r2[j] = B[(size_t)(gk + 2) * NN + cn];
            r3[j] = B[(size_t)(gk + 3) * NN + cn];
          }
        }
      }
      for (int j = 0; j < 4; ++j) {
        u32x2 t;
        t[0] = pack2bf(r0[j], r1[j]);
        t[1] = pack2bf(r2[j], r3[j]);
        *reinterpret_cast<u32x2*>(&sB[(n4 * 4 + j) * LSTR + kq * 4]) = t;
      }
    }
    if (k0 + 32 < K) {  // global_prefetch_b8 for next A tile
      int pr = m0 + (tid >> 1);
      if (!GUARD || pr < M) __builtin_prefetch(&A[(size_t)pr * K + k0 + 32], 0, 1);
    }
    __syncthreads();
    wmma_tile_step(sA, sB, wm, wn, r, hh, acc);
    __syncthreads();
  }

  // ---- K-remainder epilogue (one guarded tile; skipped when 32 | K)
  if (Kmain < K) {
    for (int idx = tid; idx < 128 * 32; idx += 256) {
      int rr = idx >> 5, cc = idx & 31;
      int gm = m0 + rr, gk = Kmain + cc;
      float v = (gm < M && gk < K) ? A[(size_t)gm * K + gk] * gate[gk] : 0.f;
      sA[rr * LSTR + cc] = f2bf(v);
    }
    for (int idx = tid; idx < 32 * 128; idx += 256) {
      int nn = idx & 127, kk = idx >> 7;
      int gk = Kmain + kk, gn = n0 + nn;
      float v = (gk < K && gn < NN) ? B[(size_t)gk * NN + gn] : 0.f;
      sB[nn * LSTR + kk] = f2bf(v);
    }
    __syncthreads();
    wmma_tile_step(sA, sB, wm, wn, r, hh, acc);
    __syncthreads();
  }

  store_tiles<GUARD>(C, M, NN, m0, n0, wm, wn, r, hh, acc);
}

__global__ __launch_bounds__(256)
void gemm_gate_wmma(const float* __restrict__ A, const float* __restrict__ gate,
                    const float* __restrict__ B, float* __restrict__ C,
                    int M, int K, int NN) {
  const int m0 = blockIdx.y * 128, n0 = blockIdx.x * 128;
  if (m0 + 128 <= M && n0 + 128 <= NN)   // block-uniform: interior fast body
    gemm1_body<false>(A, gate, B, C, M, K, NN, m0, n0);
  else
    gemm1_body<true>(A, gate, B, C, M, K, NN, m0, n0);
}

// ---------------------------------------------------------------------------
// GEMM4 body: C[M,NN] = Ab[M,K](bf16) @ W[NN,K]^T. A tile staged with
// global_load_async_to_lds_b128 (pure 16B copies); W rows are K-contiguous.
// ---------------------------------------------------------------------------
template <bool GUARD>
static __device__ inline void gemm4_body(const unsigned short* __restrict__ Ab,
                                         const float* __restrict__ W,
                                         float* __restrict__ C,
                                         int M, int K, int NN, int m0, int n0) {
  __shared__ __align__(16) unsigned short sA[128 * LSTR];
  __shared__ __align__(16) unsigned short sB[128 * LSTR];
  const int tid = threadIdx.x, lane = tid & 31, wave = tid >> 5;
  const int wm = (wave & 3) * 32, wn = (wave >> 2) * 64;
  const int r = lane & 15, hh = lane >> 4;
  v8f acc[2][4] = {};

  const int Kmain = K & ~31;
  for (int k0 = 0; k0 < Kmain; k0 += 32) {
    // ---- stage A: 128 rows x 32 k as 8-wide bf16 groups (async DMA to LDS)
    for (int i = 0; i < 2; ++i) {
      int idx = tid + i * 256;
      int rr = idx >> 2, g = idx & 3;
      int gm = m0 + rr, gk = k0 + g * 8;
      unsigned short* dstp = &sA[rr * LSTR + g * 8];
      if (!GUARD || gm < M) {
        // LDS[vdst] = MEM[vaddr] (16B per lane), tracked by ASYNCcnt.
        unsigned ldsb = (unsigned)(uintptr_t)dstp;  // low 32b of LDS aperture addr
        unsigned long long ga = (unsigned long long)(uintptr_t)(Ab + (size_t)gm * K + gk);
        asm volatile("global_load_async_to_lds_b128 %0, %1, off"
                     :: "v"(ldsb), "v"(ga) : "memory");
      } else {
        f32x4 z = {};
        *reinterpret_cast<f32x4*>(dstp) = z;
      }
    }
    // ---- stage B^T[n][k] = W[gn][gk]: 4-wide groups (1024 groups / 4 iters)
    for (int i = 0; i < 4; ++i) {
      int idx = tid + i * 256;
      int rr = idx >> 3, g = idx & 7;
      int gn = n0 + rr, gk = k0 + g * 4;
      unsigned p0 = 0, p1 = 0;
      if (!GUARD || gn < NN) {
        f32x4 v = *reinterpret_cast<const f32x4*>(W + (size_t)gn * K + gk);
        p0 = pack2bf(v[0], v[1]);
        p1 = pack2bf(v[2], v[3]);
      }
      u32x2 t; t[0] = p0; t[1] = p1;
      *reinterpret_cast<u32x2*>(&sB[rr * LSTR + g * 4]) = t;
    }
    if (k0 + 32 < K) {
      int pr = m0 + (tid >> 1);
      if (!GUARD || pr < M) __builtin_prefetch(&Ab[(size_t)pr * K + k0 + 32], 0, 1);
    }
    asm volatile("s_wait_asynccnt 0x0" ::: "memory");  // async A copies done
    __syncthreads();
    wmma_tile_step(sA, sB, wm, wn, r, hh, acc);
    __syncthreads();
  }

  if (Kmain < K) {  // K-remainder epilogue (never taken for K=512)
    for (int idx = tid; idx < 128 * 32; idx += 256) {
      int rr = idx >> 5, cc = idx & 31;
      int gm = m0 + rr, gk = Kmain + cc;
      sA[rr * LSTR + cc] = (gm < M && gk < K) ? Ab[(size_t)gm * K + gk]
                                              : (unsigned short)0;
    }
    for (int idx = tid; idx < 128 * 32; idx += 256) {
      int rr = idx >> 5, cc = idx & 31;
      int gn = n0 + rr, gk = Kmain + cc;
      float v = (gn < NN && gk < K) ? W[(size_t)gn * K + gk] : 0.f;
      sB[rr * LSTR + cc] = f2bf(v);
    }
    __syncthreads();
    wmma_tile_step(sA, sB, wm, wn, r, hh, acc);
    __syncthreads();
  }

  store_tiles<GUARD>(C, M, NN, m0, n0, wm, wn, r, hh, acc);
}

__global__ __launch_bounds__(256)
void gemm_bf16_wmma(const unsigned short* __restrict__ Ab,
                    const float* __restrict__ W, float* __restrict__ C,
                    int M, int K, int NN) {
  const int m0 = blockIdx.y * 128, n0 = blockIdx.x * 128;
  if (m0 + 128 <= M && n0 + 128 <= NN)
    gemm4_body<false>(Ab, W, C, M, K, NN, m0, n0);
  else
    gemm4_body<true>(Ab, W, C, M, K, NN, m0, n0);
}

// ---------------------------------------------------------------------------
// a_s[n] = hx1[n,:].att_src ; a_d[n] = hx1[n,:].att_dst  (one wave per node)
// ---------------------------------------------------------------------------
__global__ __launch_bounds__(256)
void alphas_kernel(const float* __restrict__ hx, const float* __restrict__ ws,
                   const float* __restrict__ wd, float* __restrict__ a_s,
                   float* __restrict__ a_d, int N) {
  int n = blockIdx.x * 8 + (threadIdx.x >> 5);
  if (n >= N) return;
  int lane = threadIdx.x & 31;
  const float* row = hx + (size_t)n * Hd;
  float ps = 0.f, pd = 0.f;
  for (int i = lane; i < Hd; i += 32) {
    float v = row[i];
    ps += v * ws[i];
    pd += v * wd[i];
  }
  for (int o = 16; o; o >>= 1) {
    ps += __shfl_down(ps, o, 32);
    pd += __shfl_down(pd, o, 32);
  }
  if (lane == 0) { a_s[n] = ps; a_d[n] = pd; }
}

// ---------------------------------------------------------------------------
// Per edge: ex = exp(sigmoid(a_s[src]+a_d[dst])); denom[dst] += ex (atomic).
// ---------------------------------------------------------------------------
__global__ __launch_bounds__(256)
void edge_exp_kernel(const int* __restrict__ src, const int* __restrict__ dst,
                     const float* __restrict__ a_s, const float* __restrict__ a_d,
                     float* __restrict__ exb, float* __restrict__ denom, int E) {
  int e = blockIdx.x * blockDim.x + threadIdx.x;
  if (e >= E) return;
  float x = a_s[src[e]] + a_d[dst[e]];
  float sg = 1.f / (1.f + expf(-x));
  float ex = expf(sg);
  exb[e] = ex;
  atomicAdd(&denom[dst[e]], ex);
}

// ---------------------------------------------------------------------------
// agg[dst,:] += hx[src,:] * (ex/denom[dst])  (one wave per edge, float4 gathers)
// ---------------------------------------------------------------------------
__global__ __launch_bounds__(256)
void edge_agg_kernel(const int* __restrict__ src, const int* __restrict__ dst,
                     const float* __restrict__ exb, const float* __restrict__ denom,
                     const float* __restrict__ hx, float* __restrict__ agg, int E) {
  int e = blockIdx.x * 8 + (threadIdx.x >> 5);
  if (e >= E) return;
  int lane = threadIdx.x & 31;
  int s = src[e], d = dst[e];
  float wg = exb[e] / (denom[d] + 1e-16f);
  const f32x4* rs = reinterpret_cast<const f32x4*>(hx + (size_t)s * Hd);
  float* rd = agg + (size_t)d * Hd;
  for (int rep = 0; rep < 4; ++rep) {
    int c4 = lane + rep * 32;
    f32x4 v = rs[c4];
    int c = c4 * 4;
    atomicAdd(rd + c + 0, v[0] * wg);
    atomicAdd(rd + c + 1, v[1] * wg);
    atomicAdd(rd + c + 2, v[2] * wg);
    atomicAdd(rd + c + 3, v[3] * wg);
  }
}

// ---------------------------------------------------------------------------
// y = elu(x) written as fp32 and bf16 (bf16 feeds the WMMA decoder GEMM).
// ---------------------------------------------------------------------------
__global__ __launch_bounds__(256)
void elu_cast_kernel(const float* __restrict__ in, float* __restrict__ outf,
                     unsigned short* __restrict__ outb, size_t n) {
  size_t i = (size_t)blockIdx.x * blockDim.x + threadIdx.x;
  size_t stride = (size_t)gridDim.x * blockDim.x;
  for (; i < n; i += stride) {
    float x = in[i];
    float y = x > 0.f ? x : expm1f(x);
    outf[i] = y;
    outb[i] = f2bf(y);
  }
}

// ---------------------------------------------------------------------------
// Fused middle stage per node (block of 64):
//   h2 = h1@W2 (->out), logp = log_softmax(h2@Wp+bp) (->out), hx3 = h2@W2^T.
// ---------------------------------------------------------------------------
__global__ __launch_bounds__(64)
void node_mid_kernel(const float* __restrict__ h1, const float* __restrict__ W2,
                     const float* __restrict__ Wp, const float* __restrict__ bp,
                     float* __restrict__ h2out, float* __restrict__ logpout,
                     float* __restrict__ hx3, int N) {
  int n = blockIdx.x;
  if (n >= N) return;
  __shared__ float s1[Hd];
  __shared__ float s2[Zd];
  __shared__ float sl[Pd];
  __shared__ float slse;
  const float* row = h1 + (size_t)n * Hd;
  for (int i = threadIdx.x; i < Hd; i += 64) s1[i] = row[i];
  __syncthreads();
  if (threadIdx.x < Zd) {
    float a = 0.f;
    for (int h = 0; h < Hd; ++h) a += s1[h] * W2[h * Zd + threadIdx.x];
    s2[threadIdx.x] = a;
    h2out[(size_t)n * Zd + threadIdx.x] = a;
  }
  __syncthreads();
  if (threadIdx.x < Pd) {
    float a = bp[threadIdx.x];
    for (int z = 0; z < Zd; ++z) a += s2[z] * Wp[z * Pd + threadIdx.x];
    sl[threadIdx.x] = a;
  }
  __syncthreads();
  if (threadIdx.x == 0) {
    float m = sl[0];
    for (int p = 1; p < Pd; ++p) m = fmaxf(m, sl[p]);
    float s = 0.f;
    for (int p = 0; p < Pd; ++p) s += expf(sl[p] - m);
    slse = m + logf(s);
  }
  __syncthreads();
  if (threadIdx.x < Pd)
    logpout[(size_t)n * Pd + threadIdx.x] = sl[threadIdx.x] - slse;
  for (int h = threadIdx.x; h < Hd; h += 64) {
    float a = 0.f;
    for (int z = 0; z < Zd; ++z) a += s2[z] * W2[h * Zd + z];
    hx3[(size_t)n * Hd + h] = a;
  }
}

// ---------------------------------------------------------------------------
extern "C" void kernel_launch(void* const* d_in, const int* in_sizes, int n_in,
                              void* d_out, int out_size, void* d_ws, size_t ws_size,
                              hipStream_t stream) {
  (void)n_in; (void)out_size; (void)ws_size;
  const float* features = (const float*)d_in[0];
  const int*   eidx     = (const int*)d_in[1];
  const float* W1       = (const float*)d_in[2];
  const float* att_s    = (const float*)d_in[3];
  const float* att_d    = (const float*)d_in[4];
  const float* W2       = (const float*)d_in[5];
  const float* Wp       = (const float*)d_in[6];
  const float* bp       = (const float*)d_in[7];
  const float* gatev    = (const float*)d_in[8];

  const int N = in_sizes[0] / Gd;
  const int E = in_sizes[1] / 2;
  const int* src = eidx;
  const int* dst = eidx + E;
  const size_t NH = (size_t)N * Hd;

  // ---- carve workspace (256B aligned) ----
  char* wptr = (char*)d_ws;
  auto carve = [&](size_t bytes) -> char* {
    char* p = wptr;
    wptr += (bytes + 255) & ~(size_t)255;
    return p;
  };
  float*          hx1 = (float*)carve(NH * 4);          // hx1, later hx3
  float*          agg = (float*)carve(NH * 4);          // atomic accumulator
  float*          h1  = (float*)carve(NH * 4);          // h1 (fp32)
  unsigned short* h3b = (unsigned short*)carve(NH * 2); // bf16 h3 (junk sink earlier)
  float*          a_s = (float*)carve((size_t)N * 4);
  float*          a_d = (float*)carve((size_t)N * 4);
  float*          den = (float*)carve((size_t)N * 4);
  float*          exb = (float*)carve((size_t)E * 4);

  float* h2o = (float*)d_out;                 // [N, Zd]
  float* h4o = h2o + (size_t)N * Zd;          // [N, Gd]
  float* lpo = h4o + (size_t)N * Gd;          // [N, Pd]

  hipMemsetAsync(den, 0, (size_t)N * 4, stream);
  hipMemsetAsync(agg, 0, NH * 4, stream);

  // conv1 encoder GEMM (bf16 WMMA): hx1 = (features*gate) @ W1
  dim3 g1((Hd + 127) / 128, (N + 127) / 128);
  gemm_gate_wmma<<<g1, 256, 0, stream>>>(features, gatev, W1, hx1, N, Gd, Hd);

  alphas_kernel<<<(N + 7) / 8, 256, 0, stream>>>(hx1, att_s, att_d, a_s, a_d, N);
  edge_exp_kernel<<<(E + 255) / 256, 256, 0, stream>>>(src, dst, a_s, a_d, exb, den, E);
  edge_agg_kernel<<<(E + 7) / 8, 256, 0, stream>>>(src, dst, exb, den, hx1, agg, E);
  elu_cast_kernel<<<2048, 256, 0, stream>>>(agg, h1, h3b /*junk, rewritten later*/, NH);

  hipMemsetAsync(agg, 0, NH * 4, stream);  // reuse accumulator for conv3

  // h2 (->out), logp (->out), hx3 = h2@W2^T (reuse hx1 buffer; hx1 is dead)
  node_mid_kernel<<<N, 64, 0, stream>>>(h1, W2, Wp, bp, h2o, lpo, hx1, N);

  // conv3 reuses conv1's edge weights (tied attention)
  edge_agg_kernel<<<(E + 7) / 8, 256, 0, stream>>>(src, dst, exb, den, hx1, agg, E);
  elu_cast_kernel<<<2048, 256, 0, stream>>>(agg, h1 /*unused fp32*/, h3b, NH);

  // conv4 decoder GEMM (bf16 WMMA): h4 = h3 @ W1^T  (W1 rows already K-major)
  dim3 g4((Gd + 127) / 128, (N + 127) / 128);
  gemm_bf16_wmma<<<g4, 256, 0, stream>>>(h3b, W1, h4o, N, Hd, Gd);
}
